// RealTimeCortexGPT_16801912062749
// MI455X (gfx1250) — matrix operand order repaired and checked by reading, not compiled, not run on hardware
//
#include <hip/hip_runtime.h>
#include <hip/hip_bf16.h>

// ---------------------------------------------------------------------------
// AdaptiveMemoryBuffer.retrieve for MI455X (gfx1250, wave32, WMMA)
//
// Roofline: 256MB key stream + 128MB score spill ~= 17us @ 23.3TB/s.
// bf16 WMMA (f32 accumulate) keeps the 34.4 GFLOP GEMM at/below that floor;
// key normalization is fused into the GEMM pass so keys are read exactly once.
// K-loop is double-buffered through LDS: one workgroup barrier per 32-K step;
// all 4 B-fragments are preloaded into distinct registers so the 8 WMMAs issue
// back-to-back behind a single s_wait_dscnt.
// ---------------------------------------------------------------------------

#define BDIM 256
#define DDIM 1024
#define MDIM 65536
#define TOPK 8
#define NEG_INF (-3.0e38f)

typedef __bf16 bf16_t;
typedef bf16_t v16bf __attribute__((ext_vector_type(16)));
typedef bf16_t v8bf  __attribute__((ext_vector_type(8)));
typedef float  v8f   __attribute__((ext_vector_type(8)));

union BFrag { v16bf v; v8bf h[2]; };

// ---------------------------------------------------------------------------
// Kernel 1: normalize query rows, emit bf16 q-hat.  grid=256 (one WG per row)
// ---------------------------------------------------------------------------
__global__ __launch_bounds__(256) void k_normalize_q(const float* __restrict__ q,
                                                     bf16_t* __restrict__ qbf) {
    __shared__ float red[256];
    const int b = blockIdx.x, t = threadIdx.x;
    const float* row = q + (size_t)b * DDIM + t * 4;
    float4 v = *(const float4*)row;
    red[t] = v.x * v.x + v.y * v.y + v.z * v.z + v.w * v.w;
    __syncthreads();
    for (int s = 128; s > 0; s >>= 1) {
        if (t < s) red[t] += red[t + s];
        __syncthreads();
    }
    const float inv = 1.0f / fmaxf(sqrtf(red[0]), 1e-8f);
    bf16_t* orow = qbf + (size_t)b * DDIM + t * 4;
    orow[0] = (bf16_t)(v.x * inv);
    orow[1] = (bf16_t)(v.y * inv);
    orow[2] = (bf16_t)(v.z * inv);
    orow[3] = (bf16_t)(v.w * inv);
}

// ---------------------------------------------------------------------------
// Kernel 2: per-key scalar: decay^dt * importance * log1p(count)
// ---------------------------------------------------------------------------
__global__ __launch_bounds__(256) void k_sfac(const float* __restrict__ importance,
                                              const int* __restrict__ atimes,
                                              const int* __restrict__ acounts,
                                              const int* __restrict__ ctime,
                                              float* __restrict__ sfac) {
    const int m = blockIdx.x * 256 + threadIdx.x;
    if (m >= MDIM) return;
    const float dt = (float)(ctime[0] - atimes[m]);
    const float retention = __expf(dt * -1.00050033e-3f);   // log(0.999)
    sfac[m] = retention * importance[m] * log1pf((float)acounts[m]);
}

// ---------------------------------------------------------------------------
// Kernel 3: scores[b][m] = (qhat_b . key_m) * sfac[m] / max(||key_m||, eps)
// One WG = 64 keys x all 256 queries.  8 waves, each wave: 2 query tiles x
// 4 key subtiles = 8 C tiles (64 acc VGPRs/lane).  Keys converted f32->bf16
// while staging through a double-buffered LDS tile; ||k||^2 accumulated by
// the same loader threads (keys read from HBM exactly once).
// ---------------------------------------------------------------------------
__global__ __launch_bounds__(256) void k_gemm_scores(const bf16_t* __restrict__ qbf,
                                                     const float* __restrict__ keys,
                                                     const float* __restrict__ sfac,
                                                     float* __restrict__ scores) {
    __shared__ bf16_t kt[2][64 * 32];  // double-buffered key tile (2 x 4KB)
    __shared__ float  sq4[256];
    __shared__ float  scl[64];

    const int t    = threadIdx.x;
    const int lane = t & 31;
    const int wave = t >> 5;
    const int l15  = lane & 15;
    const int lhi  = lane >> 4;          // 0: lanes 0-15, 1: lanes 16-31
    const int mbase = blockIdx.x * 64;

    // loader mapping: 4 threads per key row, 8 contiguous f32 each
    const int r  = t >> 2;
    const int qp = t & 3;
    const float* krow = keys + (size_t)(mbase + r) * DDIM + qp * 8;
    const int ldst = r * 32 + qp * 8;    // LDS element offset for this loader
    float ssq = 0.f;

    const v8f vzero = {0.f, 0.f, 0.f, 0.f, 0.f, 0.f, 0.f, 0.f};
    v8f acc[2][4];
#pragma unroll
    for (int i = 0; i < 2; ++i)
#pragma unroll
        for (int j = 0; j < 4; ++j) acc[i][j] = vzero;

    const int qt0 = wave * 2;
    // A fragment bases (ISA 16-bit A layout: lanes<16 K-chunks {0..7,16..23},
    // lanes>=16 K-chunks {8..15,24..31})
    const bf16_t* a0base = qbf + (size_t)(qt0 * 16 + l15) * DDIM + lhi * 8;
    const bf16_t* a1base = a0base + (size_t)16 * DDIM;

    // ---- prologue: stage K-chunk 0 into buffer 0 ----
    {
        float4 f0 = *(const float4*)(krow);
        float4 f1 = *(const float4*)(krow + 4);
        ssq += f0.x * f0.x + f0.y * f0.y + f0.z * f0.z + f0.w * f0.w
             + f1.x * f1.x + f1.y * f1.y + f1.z * f1.z + f1.w * f1.w;
        v8bf pk;
        pk[0] = (bf16_t)f0.x; pk[1] = (bf16_t)f0.y;
        pk[2] = (bf16_t)f0.z; pk[3] = (bf16_t)f0.w;
        pk[4] = (bf16_t)f1.x; pk[5] = (bf16_t)f1.y;
        pk[6] = (bf16_t)f1.z; pk[7] = (bf16_t)f1.w;
        *(v8bf*)&kt[0][ldst] = pk;
        __builtin_prefetch(krow + 32, 0, 1);               // global_prefetch_b8
    }
    __syncthreads();

    // ---- main K loop: single barrier per 32-K step ----
    for (int k0 = 0; k0 < DDIM; k0 += 32) {
        const int  cur  = (k0 >> 5) & 1;
        const bool more = (k0 + 32) < DDIM;

        // issue next tile's global loads early (overlap with WMMAs below)
        float4 n0, n1;
        if (more) {
            n0 = *(const float4*)(krow + k0 + 32);
            n1 = *(const float4*)(krow + k0 + 36);
            __builtin_prefetch(krow + k0 + 64, 0, 1);      // 2 tiles ahead
        }

        // A fragments for this wave's two query tiles (L2-resident qhat)
        BFrag a0, a1;
        const bf16_t* ap0 = a0base + k0;
        const bf16_t* ap1 = a1base + k0;
        a0.h[0] = *(const v8bf*)ap0; a0.h[1] = *(const v8bf*)(ap0 + 16);
        a1.h[0] = *(const v8bf*)ap1; a1.h[1] = *(const v8bf*)(ap1 + 16);

        // preload ALL B fragments into distinct registers: one ds clause,
        // one s_wait_dscnt, then 8 back-to-back WMMAs.
        // B fragment (32x16): lane n<16 -> col n, K {0..7,16..23}; lane>=16 -> K {8..15,24..31}
        BFrag bfr[4];
#pragma unroll
        for (int st = 0; st < 4; ++st) {
            const bf16_t* bp = &kt[cur][(st * 16 + l15) * 32 + lhi * 8];
            bfr[st].h[0] = *(const v8bf*)bp;
            bfr[st].h[1] = *(const v8bf*)(bp + 16);
        }
#pragma unroll
        for (int st = 0; st < 4; ++st) {
            acc[0][st] = __builtin_amdgcn_wmma_f32_16x16x32_bf16(
                false, a0.v, false, bfr[st].v, (short)0, acc[0][st], false, false);
            acc[1][st] = __builtin_amdgcn_wmma_f32_16x16x32_bf16(
                false, a1.v, false, bfr[st].v, (short)0, acc[1][st], false, false);
        }

        // convert + stage next tile into the other buffer
        if (more) {
            ssq += n0.x * n0.x + n0.y * n0.y + n0.z * n0.z + n0.w * n0.w
                 + n1.x * n1.x + n1.y * n1.y + n1.z * n1.z + n1.w * n1.w;
            v8bf pk;
            pk[0] = (bf16_t)n0.x; pk[1] = (bf16_t)n0.y;
            pk[2] = (bf16_t)n0.z; pk[3] = (bf16_t)n0.w;
            pk[4] = (bf16_t)n1.x; pk[5] = (bf16_t)n1.y;
            pk[6] = (bf16_t)n1.z; pk[7] = (bf16_t)n1.w;
            *(v8bf*)&kt[cur ^ 1][ldst] = pk;
        }
        __syncthreads();
    }

    // --- fold ||k||^2 partials, build fused scale ---
    sq4[t] = ssq;
    __syncthreads();
    if (t < 64) {
        const float s = sq4[t * 4] + sq4[t * 4 + 1] + sq4[t * 4 + 2] + sq4[t * 4 + 3];
        scl[t] = sfac[mbase + t] / fmaxf(sqrtf(s), 1e-8f);
    }
    __syncthreads();

    // --- epilogue: scale + spill scores ---
#pragma unroll
    for (int st = 0; st < 4; ++st) {
        const float sc  = scl[st * 16 + l15];
        const int   col = mbase + st * 16 + l15;
#pragma unroll
        for (int qq = 0; qq < 2; ++qq) {
            const int qt = qt0 + qq;
#pragma unroll
            for (int j = 0; j < 8; ++j) {
                const int row = qt * 16 + j + 8 * lhi;   // ISA C/D layout
                scores[(size_t)row * MDIM + col] = acc[qq][st][j] * sc;
            }
        }
    }
}

// ---------------------------------------------------------------------------
// Kernel 4: per-row top-8 -> softmax -> weighted gather of mem_values.
// grid = 256 (one WG per batch row), 256 threads.
// ---------------------------------------------------------------------------
__global__ __launch_bounds__(256) void k_topk_combine(const float* __restrict__ scores,
                                                      const float* __restrict__ vals,
                                                      float* __restrict__ out,
                                                      float* __restrict__ conf) {
    __shared__ float cs[2048];
    __shared__ int   ci[2048];
    __shared__ float bs[256];
    __shared__ int   bp[256];
    __shared__ float w8[TOPK];
    __shared__ int   id8[TOPK];

    const int b = blockIdx.x, t = threadIdx.x;
    const float* row = scores + (size_t)b * MDIM;

    // register top-8 (sorted descending), fully unrolled / static-indexed
    float ts[TOPK]; int ti[TOPK];
#pragma unroll
    for (int j = 0; j < TOPK; ++j) { ts[j] = NEG_INF; ti[j] = 0; }

    for (int m = t; m < MDIM; m += 256) {
        const float v = row[m];
        if (v > ts[TOPK - 1]) {
            int cnt = 0;
#pragma unroll
            for (int j = 0; j < TOPK; ++j) cnt += (ts[j] >= v) ? 1 : 0;
#pragma unroll
            for (int j = TOPK - 1; j > 0; --j)
                if (j > cnt) { ts[j] = ts[j - 1]; ti[j] = ti[j - 1]; }
#pragma unroll
            for (int j = 0; j < TOPK; ++j)
                if (j == cnt) { ts[j] = v; ti[j] = m; }
        }
    }
#pragma unroll
    for (int j = 0; j < TOPK; ++j) { cs[t * TOPK + j] = ts[j]; ci[t * TOPK + j] = ti[j]; }
    __syncthreads();

    // global top-8 via 8 rounds of LDS argmax reduction
    for (int sel = 0; sel < TOPK; ++sel) {
        float mbest = NEG_INF; int pbest = t * TOPK;
#pragma unroll
        for (int j = 0; j < TOPK; ++j) {
            const float c = cs[t * TOPK + j];
            if (c > mbest) { mbest = c; pbest = t * TOPK + j; }
        }
        bs[t] = mbest; bp[t] = pbest;
        __syncthreads();
        for (int s = 128; s > 0; s >>= 1) {
            if (t < s && bs[t + s] > bs[t]) { bs[t] = bs[t + s]; bp[t] = bp[t + s]; }
            __syncthreads();
        }
        if (t == 0) {
            w8[sel]  = bs[0];
            id8[sel] = ci[bp[0]];
            cs[bp[0]] = NEG_INF;
        }
        __syncthreads();
    }

    // softmax over 8 (w8 is sorted descending -> w8[0] is the max)
    if (t == 0) {
        const float mx = w8[0];
        float e[TOPK], sum = 0.f;
#pragma unroll
        for (int j = 0; j < TOPK; ++j) { e[j] = __expf(w8[j] - mx); sum += e[j]; }
        const float inv = 1.0f / sum;
#pragma unroll
        for (int j = 0; j < TOPK; ++j) w8[j] = e[j] * inv;
        conf[b] = w8[0];                 // max weight == exp(0)/sum
    }
    __syncthreads();

    // weighted gather-combine: each thread owns 4 contiguous d's
    float4 a = {0.f, 0.f, 0.f, 0.f};
#pragma unroll
    for (int j = 0; j < TOPK; ++j) {
        const float w = w8[j];
        const float4 v = *(const float4*)(vals + (size_t)id8[j] * DDIM + t * 4);
        a.x += w * v.x; a.y += w * v.y; a.z += w * v.z; a.w += w * v.w;
    }
    *(float4*)(out + (size_t)b * DDIM + t * 4) = a;
}

// ---------------------------------------------------------------------------
// Launcher.  Inputs: 0=query 1=mem_keys 2=mem_values 3=importance
// 4=access_times 5=access_counts 6=current_time 7=top_k(=8, hardcoded)
// Workspace: qbf (512KB) | sfac (256KB) | scores (64MB)
// Output: combined [256*1024] ++ confidence [256]
// ---------------------------------------------------------------------------
extern "C" void kernel_launch(void* const* d_in, const int* in_sizes, int n_in,
                              void* d_out, int out_size, void* d_ws, size_t ws_size,
                              hipStream_t stream) {
    const float* q          = (const float*)d_in[0];
    const float* keys       = (const float*)d_in[1];
    const float* vals       = (const float*)d_in[2];
    const float* importance = (const float*)d_in[3];
    const int*   atimes     = (const int*)d_in[4];
    const int*   acounts    = (const int*)d_in[5];
    const int*   ctime      = (const int*)d_in[6];

    char* ws = (char*)d_ws;
    bf16_t* qbf   = (bf16_t*)(ws);
    float*  sfac  = (float*)(ws + (size_t)BDIM * DDIM * sizeof(bf16_t));            // +512KB
    float*  scores = (float*)(ws + (size_t)BDIM * DDIM * sizeof(bf16_t)
                                 + (size_t)MDIM * sizeof(float));                   // +768KB

    float* out  = (float*)d_out;
    float* conf = out + (size_t)BDIM * DDIM;

    k_normalize_q<<<BDIM, 256, 0, stream>>>(q, qbf);
    k_sfac<<<MDIM / 256, 256, 0, stream>>>(importance, atimes, acounts, ctime, sfac);
    k_gemm_scores<<<MDIM / 64, 256, 0, stream>>>(qbf, keys, sfac, scores);
    k_topk_combine<<<BDIM, 256, 0, stream>>>(scores, vals, out, conf);
}
